// MSDeformableAttention_21698174779942
// MI455X (gfx1250) — matrix operand order repaired and checked by reading, not compiled, not run on hardware
//
#include <hip/hip_runtime.h>
#include <math.h>

// ---------------------------------------------------------------------------
// MS Deformable Attention for MI455X (gfx1250, wave32)
//   Phase 0: pack [W_off|W_attn] into bf16 hi/lo in WMMA B-register order
//   Phase 1: projection GEMM (offsets + attn logits) via bf16x3 WMMA,
//            3 N-tiles per wave, vectorized A loads
//   Phase 2: softmax + bilinear sampling + weighted reduction (L2-resident)
// ---------------------------------------------------------------------------

typedef __attribute__((ext_vector_type(16))) __bf16 v16bf;
typedef __attribute__((ext_vector_type(8)))  float  v8f;

#define BS       16
#define LQ       1000
#define DMODEL   256
#define NHEADS   8
#define HDIM     32
#define NPTS     12          // 4+4+4
#define NOFF     192         // NHEADS*NPTS*2
#define NATT     96          // NHEADS*NPTS
#define NCOLS    288         // NOFF + NATT
#define NTILES_N 18          // 288/16
#define KCHUNKS  8           // 256/32
#define NT       3           // N-tiles per wave in the GEMM
#define BPACK_ELEMS (NTILES_N * KCHUNKS * 32 * 16)   // 73728 bf16 per array

// ---------------------------------------------------------------------------
// Kernel 0: pack W into bf16 hi/lo arrays laid out exactly as WMMA B operand
// registers: idx = ((tn*8 + kc)*32 + lane)*16 + e, where
//   n = tn*16 + (lane&15),  k = kc*32 + (lane>>4)*16 + e.
// Each GEMM lane then reads its 16 B elements as one contiguous 32B block.
// ---------------------------------------------------------------------------
__global__ __launch_bounds__(256) void msda_pack_w(
    const float* __restrict__ Woff,    // [256, 192]
    const float* __restrict__ Wattn,   // [256, 96]
    __bf16* __restrict__ Bhi,          // [BPACK_ELEMS]
    __bf16* __restrict__ Blo)          // [BPACK_ELEMS]
{
    const int tn   = blockIdx.x;        // 0..17
    const int kc   = threadIdx.x >> 5;  // 0..7
    const int lane = threadIdx.x & 31;
    const int half = lane >> 4;
    const int n    = tn * 16 + (lane & 15);

    // uniform per block: tiles 0..11 -> W_off, 12..17 -> W_attn
    const float* __restrict__ Wp;
    int ldb, col;
    if (tn < 12) { Wp = Woff;  ldb = NOFF; col = n; }
    else         { Wp = Wattn; ldb = NATT; col = n - NOFF; }

    const size_t base = ((size_t)(tn * KCHUNKS + kc) * 32 + lane) * 16;
    #pragma unroll
    for (int e = 0; e < 16; ++e) {
        const int k = kc * 32 + half * 16 + e;
        const float w = Wp[(size_t)k * ldb + col];
        const __bf16 hi = (__bf16)w;
        Bhi[base + e] = hi;
        Blo[base + e] = (__bf16)(w - (float)hi);
    }
}

// ---------------------------------------------------------------------------
// Kernel 1: logits[M,288] = query[M,256] @ Wpacked + bias
// One wave per (16 x 48) output strip: NT=3 adjacent 16x16 tiles share the
// converted A registers. bf16x3 split => 3 WMMAs per tile per K-chunk.
// ---------------------------------------------------------------------------
__global__ __launch_bounds__(32) void msda_proj_wmma(
    const float*  __restrict__ q,      // [M, 256]
    const __bf16* __restrict__ Bhi,    // packed B hi
    const __bf16* __restrict__ Blo,    // packed B lo
    const float*  __restrict__ boff,   // [192]
    const float*  __restrict__ battn,  // [96]
    float* __restrict__ logits)        // [M, 288]
{
    const int tg   = blockIdx.x;        // 0..5  (group of NT tiles)
    const int tm   = blockIdx.y;        // 0..M/16-1
    const int lane = threadIdx.x;
    const int half = lane >> 4;
    const int l16  = lane & 15;

    const int arow = tm * 16 + l16;
    const float* __restrict__ qrow = q + (size_t)arow * DMODEL;

    v8f acc0 = {}, acc1 = {}, acc2 = {};

    for (int kc = 0; kc < KCHUNKS; ++kc) {
        // ---- A: one contiguous run of 16 floats -> 4x float4 loads ----
        // element e: K = kc*32 + half*16 + e (contiguous, matches packed B)
        // element e in 0..15
        const float* __restrict__ ap = qrow + kc * 32 + half * 16;
        const float4 a0 = *(const float4*)(ap + 0);
        const float4 a1 = *(const float4*)(ap + 4);
        const float4 a2 = *(const float4*)(ap + 8);
        const float4 a3 = *(const float4*)(ap + 12);

        float af[16] = {a0.x, a0.y, a0.z, a0.w, a1.x, a1.y, a1.z, a1.w,
                        a2.x, a2.y, a2.z, a2.w, a3.x, a3.y, a3.z, a3.w};
        v16bf ahi, alo;
        #pragma unroll
        for (int e = 0; e < 16; ++e) {
            const __bf16 h = (__bf16)af[e];
            ahi[e] = h;
            alo[e] = (__bf16)(af[e] - (float)h);
        }

        // ---- B: packed, one 32B contiguous block per lane per tile ----
        #pragma unroll
        for (int t = 0; t < NT; ++t) {
            const int tn = tg * NT + t;
            const size_t bofs = ((size_t)(tn * KCHUNKS + kc) * 32 + lane) * 16;
            const v16bf bhi = *(const v16bf*)(Bhi + bofs);
            const v16bf blo = *(const v16bf*)(Blo + bofs);

            v8f a = (t == 0) ? acc0 : (t == 1) ? acc1 : acc2;
            a = __builtin_amdgcn_wmma_f32_16x16x32_bf16(false, ahi, false, bhi,
                                                        (short)0, a, false, false);
            a = __builtin_amdgcn_wmma_f32_16x16x32_bf16(false, ahi, false, blo,
                                                        (short)0, a, false, false);
            a = __builtin_amdgcn_wmma_f32_16x16x32_bf16(false, alo, false, bhi,
                                                        (short)0, a, false, false);
            if (t == 0) acc0 = a; else if (t == 1) acc1 = a; else acc2 = a;
        }
    }

    // ---- bias + store:  D layout: VGPR r -> M = tm*16 + r + 8*half ----
    #pragma unroll
    for (int t = 0; t < NT; ++t) {
        const int tn = tg * NT + t;
        const int n  = tn * 16 + l16;
        const float bias = (tn < 12) ? boff[n] : battn[n - NOFF];
        const v8f a = (t == 0) ? acc0 : (t == 1) ? acc1 : acc2;
        #pragma unroll
        for (int r = 0; r < 8; ++r) {
            const int m = tm * 16 + r + half * 8;
            logits[(size_t)m * NCOLS + n] = a[r] + bias;
        }
    }
}

// ---------------------------------------------------------------------------
// Kernel 2: per (b,q) block of 256 threads = 8 waves (one head per wave),
// lane = channel. Softmax over 12 points, bilinear-sample 3 pyramid levels
// (values are L2-resident: 138MB < 192MB), weighted sum -> out[b,q,h*32+c].
// ---------------------------------------------------------------------------
__global__ __launch_bounds__(256) void msda_sample(
    const float* __restrict__ logits,  // [M, 288]
    const float* __restrict__ ref_pts, // [M, 4] (cx, cy, sw, sh)
    const float* __restrict__ v0,      // [bs*H, 32, 80, 80]
    const float* __restrict__ v1,      // [bs*H, 32, 40, 40]
    const float* __restrict__ v2,      // [bs*H, 32, 20, 20]
    float* __restrict__ out)           // [M, 256]
{
    const int bq = blockIdx.x;          // 0..BS*LQ-1
    const int b  = bq / LQ;
    const int h  = threadIdx.x >> 5;    // head 0..7
    const int c  = threadIdx.x & 31;    // channel 0..31

    const float* lrow = logits + (size_t)bq * NCOLS;

    // ---- softmax over the 12 attention logits of this head ----
    float aw[NPTS];
    float mx = -INFINITY;
    #pragma unroll
    for (int p = 0; p < NPTS; ++p) {
        aw[p] = lrow[NOFF + h * NPTS + p];
        mx = fmaxf(mx, aw[p]);
    }
    float ssum = 0.0f;
    #pragma unroll
    for (int p = 0; p < NPTS; ++p) {
        aw[p] = __expf(aw[p] - mx);
        ssum += aw[p];
    }
    const float inv = 1.0f / ssum;

    const float4 rp = ((const float4*)ref_pts)[bq];  // cx, cy, sw, sh
    const int    bh = b * NHEADS + h;

    float acc = 0.0f;

    #pragma unroll
    for (int p = 0; p < NPTS; ++p) {
        const int lvl = p >> 2;
        const float* __restrict__ V;
        int W, H;
        if      (lvl == 0) { V = v0; W = 80; H = 80; }
        else if (lvl == 1) { V = v1; W = 40; H = 40; }
        else               { V = v2; W = 20; H = 20; }

        const float* __restrict__ plane =
            V + ((size_t)bh * HDIM + c) * (size_t)(W * H);

        const float ox = lrow[h * (NPTS * 2) + p * 2 + 0];
        const float oy = lrow[h * (NPTS * 2) + p * 2 + 1];

        // loc = center + off * (1/4) * size * OFFSET_SCALE(0.5)
        const float locx = rp.x + ox * 0.125f * rp.z;
        const float locy = rp.y + oy * 0.125f * rp.w;

        // grid = 2*loc-1; pixel = ((grid+1)*dim - 1)*0.5 = loc*dim - 0.5
        const float x = locx * (float)W - 0.5f;
        const float y = locy * (float)H - 0.5f;

        const float fx0 = floorf(x), fy0 = floorf(y);
        const int   x0 = (int)fx0,   y0 = (int)fy0;
        const int   x1 = x0 + 1,     y1 = y0 + 1;
        const float wx1 = x - fx0, wx0 = 1.0f - wx1;
        const float wy1 = y - fy0, wy0 = 1.0f - wy1;

        float s = 0.0f;
        if (x0 >= 0 && x0 < W && y0 >= 0 && y0 < H)
            s += plane[y0 * W + x0] * (wx0 * wy0);
        if (x1 >= 0 && x1 < W && y0 >= 0 && y0 < H)
            s += plane[y0 * W + x1] * (wx1 * wy0);
        if (x0 >= 0 && x0 < W && y1 >= 0 && y1 < H)
            s += plane[y1 * W + x0] * (wx0 * wy1);
        if (x1 >= 0 && x1 < W && y1 >= 0 && y1 < H)
            s += plane[y1 * W + x1] * (wx1 * wy1);

        acc += s * (aw[p] * inv);
    }

    out[(size_t)bq * DMODEL + h * HDIM + c] = acc;
}

// ---------------------------------------------------------------------------
extern "C" void kernel_launch(void* const* d_in, const int* in_sizes, int n_in,
                              void* d_out, int out_size, void* d_ws, size_t ws_size,
                              hipStream_t stream) {
    const float* query   = (const float*)d_in[0];  // [16,1000,256]
    const float* ref_pts = (const float*)d_in[1];  // [16,1000,1,4]
    const float* value0  = (const float*)d_in[2];  // [128,32,80,80]
    const float* value1  = (const float*)d_in[3];  // [128,32,40,40]
    const float* value2  = (const float*)d_in[4];  // [128,32,20,20]
    const float* W_off   = (const float*)d_in[5];  // [256,192]
    const float* b_off   = (const float*)d_in[6];  // [192]
    const float* W_attn  = (const float*)d_in[7];  // [256,96]
    const float* b_attn  = (const float*)d_in[8];  // [96]
    float*       out     = (float*)d_out;

    const int M = in_sizes[0] / DMODEL;            // bs*Lq = 16000

    // ---- workspace layout ----
    char* ws = (char*)d_ws;
    float*  logits = (float*)ws;                               // M*288 f32
    size_t  ofs = (size_t)M * NCOLS * sizeof(float);
    ofs = (ofs + 255) & ~(size_t)255;
    __bf16* Bhi = (__bf16*)(ws + ofs);                         // 144 KB
    ofs += BPACK_ELEMS * sizeof(__bf16);
    ofs = (ofs + 255) & ~(size_t)255;
    __bf16* Blo = (__bf16*)(ws + ofs);                         // 144 KB

    // Phase 0: pack weights into WMMA B-layout bf16 hi/lo (tiny, runs once
    // per launch; deterministic).
    msda_pack_w<<<dim3(NTILES_N), dim3(256), 0, stream>>>(W_off, W_attn, Bhi, Blo);

    // Phase 1: projection GEMM, one wave per 16x48 strip
    dim3 g1(NTILES_N / NT, M / 16);
    msda_proj_wmma<<<g1, dim3(32), 0, stream>>>(query, Bhi, Blo,
                                                b_off, b_attn, logits);

    // Phase 2: softmax + bilinear sampling + weighted sum
    msda_sample<<<dim3(M), dim3(256), 0, stream>>>(logits, ref_pts,
                                                   value0, value1, value2, out);
}
